// Hybrid_10213432230045
// MI455X (gfx1250) — compile-verified, matrix operand order checked by reading, not run on hardware
//
#include <hip/hip_runtime.h>
#include <math.h>

#define N_NODES 50000
#define N_EDGES 800000
#define N_ESL   (N_NODES + N_EDGES)   // GATv2 adds self-loops
#define NUM_IN  128
#define HID     32
#define NHEAD   8
#define HC      256                    // NHEAD * 32 (HID == OUT == 32)

typedef __attribute__((ext_vector_type(8)))  __bf16 v8bf;
typedef __attribute__((ext_vector_type(16))) __bf16 v16bf;
typedef __attribute__((ext_vector_type(8)))  float  v8f;

// ---------- helpers ----------
__device__ __forceinline__ __bf16 f2bf(float f) {
  union { float f; unsigned u; } in; in.f = f;
  unsigned u = in.u;
  u += 0x7FFFu + ((u >> 16) & 1u);           // round-to-nearest-even
  union { unsigned short s; __bf16 b; } out;
  out.s = (unsigned short)(u >> 16);
  return out.b;
}

// float atomic max via monotone sign-split trick (init with -inf)
__device__ __forceinline__ void atomicMaxF(float* addr, float val) {
  if (val >= 0.0f) atomicMax((int*)addr, __float_as_int(val));
  else             atomicMin((unsigned int*)addr, __float_as_uint(val));
}

#define GS_LOOP(i, n)                                                          \
  for (long long i = (long long)blockIdx.x * blockDim.x + threadIdx.x;         \
       i < (long long)(n); i += (long long)gridDim.x * blockDim.x)

// ---------- elementwise ----------
__global__ void k_fill(float* __restrict__ p, float v, long long n) {
  GS_LOOP(i, n) p[i] = v;
}

__global__ void k_fixup(float* __restrict__ p, long long n) {  // !finite -> 0
  GS_LOOP(i, n) { float v = p[i]; if (!(fabsf(v) < INFINITY)) p[i] = 0.0f; }
}

__global__ void k_elu_bias(float* __restrict__ p, const float* __restrict__ bias,
                           long long n, int hc) {
  GS_LOOP(i, n) {
    float v = p[i] + bias[(int)(i % hc)];
    p[i] = v > 0.0f ? v : expm1f(v);
  }
}

// f32 -> bf16 convert (activations)
__global__ void k_f2bf(const float* __restrict__ s, __bf16* __restrict__ d, long long n) {
  GS_LOOP(i, n) d[i] = f2bf(s[i]);
}

// weight [K,Nc] f32 -> transposed [Nc,K] bf16 (makes B WMMA fragments contiguous)
__global__ void k_w2bft(const float* __restrict__ W, __bf16* __restrict__ Wt,
                        int K, int Nc) {
  GS_LOOP(i, (long long)K * Nc) {
    int k = (int)(i / Nc), n = (int)(i - (long long)k * Nc);
    Wt[(long long)n * K + k] = f2bf(W[i]);
  }
}

// ---------- GraphConv scatter-max: agg[dst,f] = max(x[src,f] * ew[e]) ----------
__global__ void k_scatter_max(const float* __restrict__ x, const int* __restrict__ src,
                              const int* __restrict__ dst, const float* __restrict__ ew,
                              float* __restrict__ agg, int F) {
  GS_LOOP(i, (long long)N_EDGES * F) {
    int e = (int)(i / F);
    int f = (int)(i - (long long)e * F);
    float v = x[(long long)src[e] * F + f] * ew[e];
    atomicMaxF(&agg[(long long)dst[e] * F + f], v);
  }
}

// ---------- GATv2 edge passes (C = 32 channels / head in both layers) ----------
__global__ void k_gat_score(const float* __restrict__ xl, const float* __restrict__ xr,
                            const float* __restrict__ att, const int* __restrict__ src,
                            const int* __restrict__ dst, float* __restrict__ score,
                            float* __restrict__ m) {
  GS_LOOP(i, (long long)N_ESL * NHEAD) {
    long long e = i >> 3; int h = (int)(i & 7);
    int s, d;
    if (e < N_EDGES) { s = src[e]; d = dst[e]; } else { s = d = (int)(e - N_EDGES); }
    const float* pl = xl + (long long)s * HC + h * 32;
    const float* pr = xr + (long long)d * HC + h * 32;
    const float* pa = att + h * 32;
    float sc = 0.0f;
#pragma unroll
    for (int c = 0; c < 32; ++c) {
      float v = pl[c] + pr[c];
      v = v > 0.0f ? v : 0.2f * v;           // leaky relu
      sc = fmaf(v, pa[c], sc);
    }
    score[i] = sc;
    atomicMaxF(&m[(long long)d * NHEAD + h], sc);
  }
}

__global__ void k_gat_exp(float* __restrict__ score, const float* __restrict__ m,
                          float* __restrict__ den, const int* __restrict__ dst) {
  GS_LOOP(i, (long long)N_ESL * NHEAD) {
    long long e = i >> 3; int h = (int)(i & 7);
    int d = (e < N_EDGES) ? dst[e] : (int)(e - N_EDGES);
    float a = __expf(score[i] - m[(long long)d * NHEAD + h]);
    score[i] = a;
    atomicAdd(&den[(long long)d * NHEAD + h], a);
  }
}

__global__ void k_gat_scatter(const float* __restrict__ xl, const float* __restrict__ score,
                              const float* __restrict__ den, const int* __restrict__ src,
                              const int* __restrict__ dst, float* __restrict__ gout) {
  GS_LOOP(i, (long long)N_ESL * NHEAD) {
    long long e = i >> 3; int h = (int)(i & 7);
    int s, d;
    if (e < N_EDGES) { s = src[e]; d = dst[e]; } else { s = d = (int)(e - N_EDGES); }
    float alpha = score[i] / (den[(long long)d * NHEAD + h] + 1e-16f);
    const float* pl = xl + (long long)s * HC + h * 32;
    float* po = gout + (long long)d * HC + h * 32;
#pragma unroll
    for (int c = 0; c < 32; ++c) atomicAdd(&po[c], alpha * pl[c]);
  }
}

// ---------- bf16 WMMA GEMM: C = act(A1@B1 [+ A2@B2] + bias) ----------
// A: [M,K] bf16 row-major; Bt: [Nc,K] bf16 (pre-transposed weight).
// One wave computes a 16 x (16*STRIP) strip: A fragment reused STRIP times.
// All fragments are contiguous 16B loads (global_load_b128 of bf16).
template <int STRIP>
__global__ void k_wmma_gemm(const __bf16* __restrict__ A1, const __bf16* __restrict__ Bt1,
                            const __bf16* __restrict__ A2, const __bf16* __restrict__ Bt2,
                            const float* __restrict__ bias, float* __restrict__ C,
                            int M, int Nc, int K, int act) {
  const int lane   = threadIdx.x & 31;
  const int wave   = threadIdx.x >> 5;
  const int stripsN = Nc / (16 * STRIP);
  const int strip  = blockIdx.x * (blockDim.x >> 5) + wave;
  if (strip >= (M >> 4) * stripsN) return;        // wave-uniform: EXEC stays all-1s
  const int tm = strip / stripsN;
  const int ts = strip - tm * stripsN;
  const int hi  = lane >> 4;                      // lane half -> K offset +8
  const int l15 = lane & 15;
  const int nbase = ts * (16 * STRIP);
  const long long arowoff = (long long)(tm * 16 + l15) * K;

  v8f acc[STRIP];
#pragma unroll
  for (int t = 0; t < STRIP; ++t) {
    const float bv = bias ? bias[nbase + t * 16 + l15] : 0.0f;
#pragma unroll
    for (int r = 0; r < 8; ++r) acc[t][r] = bv;
  }

  const __bf16* Ap = A1;
  const __bf16* Bp = Bt1;
  for (int pass = 0; pass < 2 && Ap; ++pass) {
    const __bf16* Arow = Ap + arowoff;
    for (int kk = 0; kk < K; kk += 32) {
      const int k0 = kk + (hi << 3);              // elems 0..7 contiguous
      const int k1 = k0 + 16;                     // elems 8..15 contiguous
      if (kk + 32 < K) __builtin_prefetch(Arow + kk + 32, 0, 3);
      v8bf alo = *(const v8bf*)(Arow + k0);
      v8bf ahi = *(const v8bf*)(Arow + k1);
      v16bf a = __builtin_shufflevector(alo, ahi, 0, 1, 2, 3, 4, 5, 6, 7,
                                        8, 9, 10, 11, 12, 13, 14, 15);
#pragma unroll
      for (int t = 0; t < STRIP; ++t) {
        const __bf16* Bcol = Bp + (long long)(nbase + t * 16 + l15) * K;
        v8bf blo = *(const v8bf*)(Bcol + k0);
        v8bf bhi = *(const v8bf*)(Bcol + k1);
        v16bf b = __builtin_shufflevector(blo, bhi, 0, 1, 2, 3, 4, 5, 6, 7,
                                          8, 9, 10, 11, 12, 13, 14, 15);
        acc[t] = __builtin_amdgcn_wmma_f32_16x16x32_bf16(false, a, false, b,
                                                         (short)0, acc[t], false, false);
      }
    }
    Ap = A2; Bp = Bt2;
  }

#pragma unroll
  for (int t = 0; t < STRIP; ++t)
#pragma unroll
    for (int r = 0; r < 8; ++r) {                 // C layout: vgpr r -> row r + 8*hi
      float v = acc[t][r];
      if (act == 1) v = fmaxf(v, 0.0f);
      C[(long long)(tm * 16 + hi * 8 + r) * Nc + nbase + t * 16 + l15] = v;
    }
}

static inline void gemm(const __bf16* A1, const __bf16* B1, const __bf16* A2,
                        const __bf16* B2, const float* bias, float* C,
                        int M, int Nc, int K, int act, hipStream_t s) {
  if (Nc % 64 == 0) {
    int strips = (M / 16) * (Nc / 64);
    k_wmma_gemm<4><<<(strips + 3) / 4, 128, 0, s>>>(A1, B1, A2, B2, bias, C, M, Nc, K, act);
  } else {
    int strips = (M / 16) * (Nc / 32);
    k_wmma_gemm<2><<<(strips + 3) / 4, 128, 0, s>>>(A1, B1, A2, B2, bias, C, M, Nc, K, act);
  }
}

// ---------- orchestration ----------
extern "C" void kernel_launch(void* const* d_in, const int* in_sizes, int n_in,
                              void* d_out, int out_size, void* d_ws, size_t ws_size,
                              hipStream_t stream) {
  (void)in_sizes; (void)n_in; (void)out_size; (void)ws_size;
  const float* x       = (const float*)d_in[0];
  const int*   ei      = (const int*)d_in[1];
  const float* ew      = (const float*)d_in[2];
  const float* W_rel1  = (const float*)d_in[3];
  const float* b_rel1  = (const float*)d_in[4];
  const float* W_root1 = (const float*)d_in[5];
  const float* W_rel2  = (const float*)d_in[6];
  const float* b_rel2  = (const float*)d_in[7];
  const float* W_root2 = (const float*)d_in[8];
  const float* Wl1     = (const float*)d_in[9];
  const float* bl1     = (const float*)d_in[10];
  const float* Wr1     = (const float*)d_in[11];
  const float* br1     = (const float*)d_in[12];
  const float* att1    = (const float*)d_in[13];
  const float* bias1   = (const float*)d_in[14];
  const float* Wl2     = (const float*)d_in[15];
  const float* bl2     = (const float*)d_in[16];
  const float* Wr2     = (const float*)d_in[17];
  const float* br2     = (const float*)d_in[18];
  const float* att2    = (const float*)d_in[19];
  const float* bias2   = (const float*)d_in[20];
  const float* W_lin   = (const float*)d_in[21];
  const float* b_lin   = (const float*)d_in[22];
  float* out = (float*)d_out;

  const int* src = ei;             // edge_index[0]
  const int* dst = ei + N_EDGES;   // edge_index[1]

  // ---- workspace carve-up: f32 region then bf16 region ----
  float* ws  = (float*)d_ws;
  size_t o   = 0;
  float* agg = ws + o; o += (size_t)N_NODES * NUM_IN;   // reused for both GraphConvs
  float* h1  = ws + o; o += (size_t)N_NODES * HID;
  float* h2  = ws + o; o += (size_t)N_NODES * HID;
  float* xl  = ws + o; o += (size_t)N_NODES * HC;       // reused by both GAT layers
  float* xr  = ws + o; o += (size_t)N_NODES * HC;
  float* g1  = ws + o; o += (size_t)N_NODES * HC;
  float* g2  = ws + o; o += (size_t)N_NODES * HC;
  float* mb  = ws + o; o += (size_t)N_NODES * NHEAD;
  float* den = ws + o; o += (size_t)N_NODES * NHEAD;
  float* sc  = ws + o; o += (size_t)N_ESL * NHEAD;

  __bf16* bws = (__bf16*)(ws + o);
  size_t b = 0;
  __bf16* abf1     = bws + b; b += (size_t)N_NODES * HC;      // activations (<= K=256)
  __bf16* abf2     = bws + b; b += (size_t)N_NODES * NUM_IN;  // 2nd operand (<= K=128)
  __bf16* tw_rel1  = bws + b; b += NUM_IN * HID;
  __bf16* tw_root1 = bws + b; b += NUM_IN * HID;
  __bf16* tw_rel2  = bws + b; b += HID * HID;
  __bf16* tw_root2 = bws + b; b += HID * HID;
  __bf16* twl1     = bws + b; b += HID * HC;
  __bf16* twr1     = bws + b; b += HID * HC;
  __bf16* twl2     = bws + b; b += HC * HC;
  __bf16* twr2     = bws + b; b += HC * HC;
  __bf16* twlin    = bws + b; b += HC * 32;

  const dim3 EB(4096), ET(256), WB(64);
  const float NINF = -INFINITY;

  // ---- weight prep: f32 [K,Nc] -> bf16 [Nc,K] ----
  k_w2bft<<<WB, ET, 0, stream>>>(W_rel1,  tw_rel1,  NUM_IN, HID);
  k_w2bft<<<WB, ET, 0, stream>>>(W_root1, tw_root1, NUM_IN, HID);
  k_w2bft<<<WB, ET, 0, stream>>>(W_rel2,  tw_rel2,  HID, HID);
  k_w2bft<<<WB, ET, 0, stream>>>(W_root2, tw_root2, HID, HID);
  k_w2bft<<<WB, ET, 0, stream>>>(Wl1,     twl1,     HID, HC);
  k_w2bft<<<WB, ET, 0, stream>>>(Wr1,     twr1,     HID, HC);
  k_w2bft<<<WB, ET, 0, stream>>>(Wl2,     twl2,     HC, HC);
  k_w2bft<<<WB, ET, 0, stream>>>(Wr2,     twr2,     HC, HC);
  k_w2bft<<<WB, ET, 0, stream>>>(W_lin,   twlin,    HC, 32);

  // ---- GraphConv 1: relu(lin_rel(max_agg(x*ew)) + lin_root(x)) ----
  k_fill<<<EB, ET, 0, stream>>>(agg, NINF, (long long)N_NODES * NUM_IN);
  k_scatter_max<<<EB, ET, 0, stream>>>(x, src, dst, ew, agg, NUM_IN);
  k_fixup<<<EB, ET, 0, stream>>>(agg, (long long)N_NODES * NUM_IN);
  k_f2bf<<<EB, ET, 0, stream>>>(agg, abf1, (long long)N_NODES * NUM_IN);
  k_f2bf<<<EB, ET, 0, stream>>>(x,   abf2, (long long)N_NODES * NUM_IN);
  gemm(abf1, tw_rel1, abf2, tw_root1, b_rel1, h1, N_NODES, HID, NUM_IN, 1, stream);

  // ---- GraphConv 2 ----
  k_fill<<<EB, ET, 0, stream>>>(agg, NINF, (long long)N_NODES * HID);
  k_scatter_max<<<EB, ET, 0, stream>>>(h1, src, dst, ew, agg, HID);
  k_fixup<<<EB, ET, 0, stream>>>(agg, (long long)N_NODES * HID);
  k_f2bf<<<EB, ET, 0, stream>>>(agg, abf1, (long long)N_NODES * HID);
  k_f2bf<<<EB, ET, 0, stream>>>(h1,  abf2, (long long)N_NODES * HID);
  gemm(abf1, tw_rel2, abf2, tw_root2, b_rel2, h2, N_NODES, HID, HID, 1, stream);

  // ---- GATv2 layer 1 ----
  k_f2bf<<<EB, ET, 0, stream>>>(h2, abf1, (long long)N_NODES * HID);
  gemm(abf1, twl1, nullptr, nullptr, bl1, xl, N_NODES, HC, HID, 0, stream);
  gemm(abf1, twr1, nullptr, nullptr, br1, xr, N_NODES, HC, HID, 0, stream);
  k_fill<<<EB, ET, 0, stream>>>(mb, NINF, (long long)N_NODES * NHEAD);
  k_gat_score<<<EB, ET, 0, stream>>>(xl, xr, att1, src, dst, sc, mb);
  k_fill<<<EB, ET, 0, stream>>>(den, 0.0f, (long long)N_NODES * NHEAD);
  k_gat_exp<<<EB, ET, 0, stream>>>(sc, mb, den, dst);
  k_fill<<<EB, ET, 0, stream>>>(g1, 0.0f, (long long)N_NODES * HC);
  k_gat_scatter<<<EB, ET, 0, stream>>>(xl, sc, den, src, dst, g1);
  k_elu_bias<<<EB, ET, 0, stream>>>(g1, bias1, (long long)N_NODES * HC, HC);

  // ---- GATv2 layer 2 ----
  k_f2bf<<<EB, ET, 0, stream>>>(g1, abf1, (long long)N_NODES * HC);
  gemm(abf1, twl2, nullptr, nullptr, bl2, xl, N_NODES, HC, HC, 0, stream);
  gemm(abf1, twr2, nullptr, nullptr, br2, xr, N_NODES, HC, HC, 0, stream);
  k_fill<<<EB, ET, 0, stream>>>(mb, NINF, (long long)N_NODES * NHEAD);
  k_gat_score<<<EB, ET, 0, stream>>>(xl, xr, att2, src, dst, sc, mb);
  k_fill<<<EB, ET, 0, stream>>>(den, 0.0f, (long long)N_NODES * NHEAD);
  k_gat_exp<<<EB, ET, 0, stream>>>(sc, mb, den, dst);
  k_fill<<<EB, ET, 0, stream>>>(g2, 0.0f, (long long)N_NODES * HC);
  k_gat_scatter<<<EB, ET, 0, stream>>>(xl, sc, den, src, dst, g2);
  k_elu_bias<<<EB, ET, 0, stream>>>(g2, bias2, (long long)N_NODES * HC, HC);

  // ---- final linear ----
  k_f2bf<<<EB, ET, 0, stream>>>(g2, abf1, (long long)N_NODES * HC);
  gemm(abf1, twlin, nullptr, nullptr, b_lin, out, N_NODES, 32, HC, 0, stream);
}
